// PrismDecoder_88416196756048
// MI455X (gfx1250) — matrix operand order repaired and checked by reading, not compiled, not run on hardware
//
#include <hip/hip_runtime.h>

typedef __bf16 bf16_t;
typedef __attribute__((ext_vector_type(16))) __bf16 v16bf;
typedef __attribute__((ext_vector_type(8)))  __bf16 v8bf;
typedef __attribute__((ext_vector_type(8)))  float  v8f;
typedef __attribute__((ext_vector_type(4)))  int    v4i;
typedef __attribute__((address_space(1)))    v4i    v4i_g;   // global
typedef __attribute__((address_space(3)))    v4i    v4i_l;   // LDS

#define N_VERTS 100000
#define N_FACES 200000
#define DFEAT   512

// ---- CDNA5 async global->LDS staging (guarded; falls back to reg-staged copy) ----
#if defined(__has_builtin)
#if __has_builtin(__builtin_amdgcn_global_load_async_to_lds_b128) && \
    __has_builtin(__builtin_amdgcn_s_wait_asynccnt)
#define HAVE_ASYNC_LDS 1
#endif
#endif
#ifndef HAVE_ASYNC_LDS
#define HAVE_ASYNC_LDS 0
#endif

__device__ __forceinline__ void copy16B_g2l(bf16_t* lds_dst, const bf16_t* gsrc) {
#if HAVE_ASYNC_LDS
  void* gp = (void*)gsrc;      // drop const, generic
  void* lp = (void*)lds_dst;   // generic (from __shared__)
  __builtin_amdgcn_global_load_async_to_lds_b128((v4i_g*)gp, (v4i_l*)lp, 0, 0);
#else
  *(uint4*)lds_dst = *(const uint4*)gsrc;
#endif
}

__device__ __forceinline__ void wait_async_lds() {
#if HAVE_ASYNC_LDS
  __builtin_amdgcn_s_wait_asynccnt(0);
#endif
}

// -------------------- weight convert + transpose: Wt[n*K + k] = bf16(W[k*N + n]) --------
__global__ __launch_bounds__(256)
void cvt_transpose_kernel(const float* __restrict__ src, bf16_t* __restrict__ dst,
                          int K, int N) {
  const int idx = blockIdx.x * 256 + threadIdx.x;   // coalesced read of src
  if (idx >= K * N) return;
  const int k = idx / N;
  const int n = idx - k * N;
  dst[(size_t)n * K + k] = (bf16_t)src[idx];
}

// -------------------- gather features[faces] and mean over corners --------------------
__global__ __launch_bounds__(128)
void gather_mean_kernel(const float* __restrict__ features, const int* __restrict__ faces,
                        bf16_t* __restrict__ out) {
  const int f  = blockIdx.x;
  const int i0 = faces[3 * f + 0];
  const int i1 = faces[3 * f + 1];
  const int i2 = faces[3 * f + 2];
  const int d  = threadIdx.x * 4;
  const float4 a = *(const float4*)(features + (size_t)i0 * DFEAT + d);
  const float4 b = *(const float4*)(features + (size_t)i1 * DFEAT + d);
  const float4 c = *(const float4*)(features + (size_t)i2 * DFEAT + d);
  const float s = 1.0f / 3.0f;
  bf16_t* dst = out + (size_t)f * DFEAT + d;
  dst[0] = (bf16_t)((a.x + b.x + c.x) * s);
  dst[1] = (bf16_t)((a.y + b.y + c.y) * s);
  dst[2] = (bf16_t)((a.z + b.z + c.z) * s);
  dst[3] = (bf16_t)((a.w + b.w + c.w) * s);
}

// -------------------- tiled bf16 WMMA GEMM: Y = relu(X @ W + bias) --------------------
// X : M x K bf16 row-major.  Wt : N x K bf16 row-major (pre-transposed weights).
// BM=64, BN=128, BK=32. 256 threads = 8 waves (wm 0..3 x wn 0..1); each wave owns a
// 16x64 strip -> 4 v_wmma_f32_16x16x32_bf16 per K-step sharing one A fragment.
// Double-buffered LDS; next tile staged (async where available) before computing.
// Requires M%64==0, N%128==0, K%32==0 (holds: M=200000, N in {512,256}, K=512).
__global__ __launch_bounds__(256)
void gemm_bf16_wmma_kernel(const bf16_t* __restrict__ X, const bf16_t* __restrict__ Wt,
                           const float* __restrict__ bias, bf16_t* __restrict__ Y,
                           int M, int N, int K, int relu) {
  constexpr int BM = 64, BN = 128, BK = 32;
  constexpr int ASZ = BM * BK, BSZ = BN * BK;
  __shared__ bf16_t As[2 * ASZ];   // row-major: As[m*BK + k]
  __shared__ bf16_t Bs[2 * BSZ];   // column-major vs output: Bs[n*BK + k] (K contiguous)

  const int tid  = threadIdx.x;
  const int wave = tid >> 5;
  const int lane = tid & 31;
  const int wm   = wave & 3;           // 0..3 : 16-row slice
  const int wn   = wave >> 2;          // 0..1 : 64-col slice
  const int h    = lane >> 4;          // lane half (ISA K-split)
  const int r    = lane & 15;
  const int bm0  = blockIdx.y * BM;
  const int bn0  = blockIdx.x * BN;

  // staging assignments (16B per copy)
  const int ar = tid >> 2, ac = (tid & 3) * 8;   // A: 64 rows x 4 thr x 8 bf16
  const int br = tid >> 1, bc = (tid & 1) * 16;  // B: 128 rows x 2 thr x 16 bf16

  v8f acc[4];
#pragma unroll
  for (int j = 0; j < 4; j++) acc[j] = (v8f){};

  // prologue: stage tile 0 into buffer 0
  copy16B_g2l(As + ar * BK + ac,      X  + (size_t)(bm0 + ar) * K + ac);
  copy16B_g2l(Bs + br * BK + bc,      Wt + (size_t)(bn0 + br) * K + bc);
  copy16B_g2l(Bs + br * BK + bc + 8,  Wt + (size_t)(bn0 + br) * K + bc + 8);
  wait_async_lds();
  __syncthreads();

  int s = 0;
  for (int k0 = 0; k0 < K; k0 += BK, s ^= 1) {
    const bf16_t* Ac = As + s * ASZ;
    const bf16_t* Bc = Bs + s * BSZ;
    const bool more = (k0 + BK) < K;
    if (more) {  // prefetch next tile into the other buffer (overlaps with WMMA below)
      bf16_t* An = As + (s ^ 1) * ASZ;
      bf16_t* Bn = Bs + (s ^ 1) * BSZ;
      const int kn = k0 + BK;
      copy16B_g2l(An + ar * BK + ac,     X  + (size_t)(bm0 + ar) * K + kn + ac);
      copy16B_g2l(Bn + br * BK + bc,     Wt + (size_t)(bn0 + br) * K + kn + bc);
      copy16B_g2l(Bn + br * BK + bc + 8, Wt + (size_t)(bn0 + br) * K + kn + bc + 8);
    }

    // A fragment: row = wm*16 + r; lane-half h holds K = [h*16, h*16+16)
    const int aro = (wm * 16 + r) * BK;
    v8bf alo = *(const v8bf*)(Ac + aro + h * 16);
    v8bf ahi = *(const v8bf*)(Ac + aro + h * 16 + 8);
    v16bf afrag = __builtin_shufflevector(alo, ahi, 0, 1, 2, 3, 4, 5, 6, 7,
                                          8, 9, 10, 11, 12, 13, 14, 15);

    // 4 B fragments: col n = wn*64 + j*16 + r, lane-half h holds K = [h*16, h*16+16)
#pragma unroll
    for (int j = 0; j < 4; j++) {
      v16bf bfrag = *(const v16bf*)(Bc + (wn * 64 + j * 16 + r) * BK + h * 16);
      acc[j] = __builtin_amdgcn_wmma_f32_16x16x32_bf16(false, afrag, false, bfrag,
                                                       (short)0, acc[j], false, false);
    }

    if (more) wait_async_lds();
    __syncthreads();
  }

  // C/D layout: VGPR i -> row M = h*8 + i, lane r -> col N = r
  const int cn0 = bn0 + wn * 64 + r;
#pragma unroll
  for (int j = 0; j < 4; j++) {
    const float bv = bias[cn0 + j * 16];
#pragma unroll
    for (int i = 0; i < 8; i++) {
      const int row = bm0 + wm * 16 + h * 8 + i;
      float v = acc[j][i] + bv;
      if (relu) v = fmaxf(v, 0.0f);
      Y[(size_t)row * N + cn0 + j * 16] = (bf16_t)v;
    }
  }
}

// -------------------- head: out12 = h3(200000x256 bf16) @ W4(256x12) + b4 --------------------
__global__ __launch_bounds__(256)
void head_kernel(const bf16_t* __restrict__ h3, const float* __restrict__ W4,
                 const float* __restrict__ b4, float* __restrict__ out12, int nFaces) {
  __shared__ float W4s[256 * 12];
  for (int i = threadIdx.x; i < 256 * 12; i += 256) W4s[i] = W4[i];
  __syncthreads();
  const int f = blockIdx.x * 256 + threadIdx.x;
  if (f >= nFaces) return;
  float acc[12];
#pragma unroll
  for (int j = 0; j < 12; j++) acc[j] = b4[j];
  const bf16_t* xr = h3 + (size_t)f * 256;
  for (int k = 0; k < 256; k++) {
    const float x = (float)xr[k];
#pragma unroll
    for (int j = 0; j < 12; j++) acc[j] = fmaf(x, W4s[k * 12 + j], acc[j]);
  }
#pragma unroll
  for (int j = 0; j < 12; j++) out12[(size_t)f * 12 + j] = acc[j];
}

// -------------------- special procrustes: nearest proper rotation --------------------
// Newton polar iteration X <- 0.5*(X + X^-T); det<0 pre-negated so result is det=+1.
__global__ __launch_bounds__(256)
void procrustes_kernel(const float* __restrict__ out12, float* __restrict__ rot, int nFaces) {
  const int f = blockIdx.x * 256 + threadIdx.x;
  if (f >= nFaces) return;
  float X[9];
  float fn = 0.0f;
#pragma unroll
  for (int i = 0; i < 9; i++) { X[i] = out12[(size_t)f * 12 + i]; fn += X[i] * X[i]; }
  const float inv = rsqrtf(fn + 1e-30f);
#pragma unroll
  for (int i = 0; i < 9; i++) X[i] *= inv;
  {
    const float d = X[0]*(X[4]*X[8]-X[5]*X[7]) - X[1]*(X[3]*X[8]-X[5]*X[6])
                  + X[2]*(X[3]*X[7]-X[4]*X[6]);
    if (d < 0.0f) {
#pragma unroll
      for (int i = 0; i < 9; i++) X[i] = -X[i];
    }
  }
  for (int it = 0; it < 12; it++) {
    float C[9];
    C[0] = X[4]*X[8]-X[5]*X[7];  C[1] = X[5]*X[6]-X[3]*X[8];  C[2] = X[3]*X[7]-X[4]*X[6];
    C[3] = X[2]*X[7]-X[1]*X[8];  C[4] = X[0]*X[8]-X[2]*X[6];  C[5] = X[1]*X[6]-X[0]*X[7];
    C[6] = X[1]*X[5]-X[2]*X[4];  C[7] = X[2]*X[3]-X[0]*X[5];  C[8] = X[0]*X[4]-X[1]*X[3];
    const float d  = X[0]*C[0] + X[1]*C[1] + X[2]*C[2];
    const float id = 1.0f / (d + copysignf(1e-12f, d));
#pragma unroll
    for (int i = 0; i < 9; i++) X[i] = 0.5f * (X[i] + C[i] * id);
  }
#pragma unroll
  for (int i = 0; i < 9; i++) rot[(size_t)f * 9 + i] = X[i];
}

// -------------------- transform prisms + scatter vertex sums --------------------
__global__ __launch_bounds__(256)
void transform_scatter_kernel(const float* __restrict__ verts, const int* __restrict__ faces,
                              const float* __restrict__ rot, const float* __restrict__ out12,
                              float* __restrict__ trans, float* __restrict__ sums,
                              float* __restrict__ counts, int nFaces) {
  const int f = blockIdx.x * 256 + threadIdx.x;
  if (f >= nFaces) return;
  float R[9];
#pragma unroll
  for (int i = 0; i < 9; i++) R[i] = rot[(size_t)f * 9 + i];
  const float t0 = out12[(size_t)f * 12 + 9];
  const float t1 = out12[(size_t)f * 12 + 10];
  const float t2 = out12[(size_t)f * 12 + 11];
#pragma unroll
  for (int v = 0; v < 3; v++) {
    const int idx = faces[3 * f + v];
    const float p0 = verts[(size_t)idx * 3 + 0];
    const float p1 = verts[(size_t)idx * 3 + 1];
    const float p2 = verts[(size_t)idx * 3 + 2];
    const float q0 = fmaf(p0, R[0], fmaf(p1, R[3], fmaf(p2, R[6], t0)));
    const float q1 = fmaf(p0, R[1], fmaf(p1, R[4], fmaf(p2, R[7], t1)));
    const float q2 = fmaf(p0, R[2], fmaf(p1, R[5], fmaf(p2, R[8], t2)));
    const size_t o = (size_t)f * 9 + (size_t)v * 3;
    trans[o + 0] = q0; trans[o + 1] = q1; trans[o + 2] = q2;
    atomicAdd(&sums[(size_t)idx * 3 + 0], q0);
    atomicAdd(&sums[(size_t)idx * 3 + 1], q1);
    atomicAdd(&sums[(size_t)idx * 3 + 2], q2);
    atomicAdd(&counts[idx], 1.0f);
  }
}

__global__ __launch_bounds__(256)
void zero_kernel(float* __restrict__ p, int n) {
  int i = blockIdx.x * 256 + threadIdx.x;
  if (i < n) p[i] = 0.0f;
}

__global__ __launch_bounds__(256)
void finalize_kernel(const float* __restrict__ sums, const float* __restrict__ counts,
                     float* __restrict__ outf, int nVerts) {
  const int v = blockIdx.x * 256 + threadIdx.x;
  if (v >= nVerts) return;
  const float c = fmaxf(counts[v], 1.0f);
  outf[(size_t)v * 3 + 0] = sums[(size_t)v * 3 + 0] / c;
  outf[(size_t)v * 3 + 1] = sums[(size_t)v * 3 + 1] / c;
  outf[(size_t)v * 3 + 2] = sums[(size_t)v * 3 + 2] / c;
}

extern "C" void kernel_launch(void* const* d_in, const int* in_sizes, int n_in,
                              void* d_out, int out_size, void* d_ws, size_t ws_size,
                              hipStream_t stream) {
  (void)in_sizes; (void)n_in; (void)out_size; (void)ws_size;
  const float* verts    = (const float*)d_in[0];
  const float* features = (const float*)d_in[1];
  const int*   faces    = (const int*)d_in[2];
  const float* W1 = (const float*)d_in[3];
  const float* b1 = (const float*)d_in[4];
  const float* W2 = (const float*)d_in[5];
  const float* b2 = (const float*)d_in[6];
  const float* W3 = (const float*)d_in[7];
  const float* b3 = (const float*)d_in[8];
  const float* W4 = (const float*)d_in[9];
  const float* b4 = (const float*)d_in[10];

  // d_out layout: out_features (1,100000,3) | transformed_prism (200000,3,3) | rotations (200000,3,3)
  float* out_features = (float*)d_out;
  float* trans_out    = out_features + (size_t)N_VERTS * 3;
  float* rot_out      = trans_out + (size_t)N_FACES * 9;

  // workspace carve-out (256B aligned blocks)
  char* ws = (char*)d_ws;
  size_t off = 0;
  auto carve = [&](size_t bytes) {
    void* p = ws + off;
    off += (bytes + 255) & ~(size_t)255;
    return p;
  };
  bf16_t* bufA   = (bf16_t*)carve((size_t)N_FACES * DFEAT * 2);  // face_feat -> h2
  bf16_t* bufB   = (bf16_t*)carve((size_t)N_FACES * DFEAT * 2);  // h1 -> h3
  bf16_t* W1t    = (bf16_t*)carve((size_t)512 * 512 * 2);        // N x K (transposed)
  bf16_t* W2t    = (bf16_t*)carve((size_t)512 * 512 * 2);
  bf16_t* W3t    = (bf16_t*)carve((size_t)256 * 512 * 2);
  float*  out12  = (float*)carve((size_t)N_FACES * 12 * 4);
  float*  sums   = (float*)carve((size_t)N_VERTS * 3 * 4);
  float*  counts = (float*)carve((size_t)N_VERTS * 4);

  // weights -> bf16, transposed to N x K so GEMM B tiles are K-contiguous
  cvt_transpose_kernel<<<(512 * 512 + 255) / 256, 256, 0, stream>>>(W1, W1t, 512, 512);
  cvt_transpose_kernel<<<(512 * 512 + 255) / 256, 256, 0, stream>>>(W2, W2t, 512, 512);
  cvt_transpose_kernel<<<(512 * 256 + 255) / 256, 256, 0, stream>>>(W3, W3t, 512, 256);

  // gather + mean -> bf16 activations
  gather_mean_kernel<<<N_FACES, 128, 0, stream>>>(features, faces, bufA);

  // MLP on the matrix cores (bf16 WMMA, f32 accumulate)
  dim3 g1(512 / 128, N_FACES / 64);
  gemm_bf16_wmma_kernel<<<g1, 256, 0, stream>>>(bufA, W1t, b1, bufB, N_FACES, 512, 512, 1);
  gemm_bf16_wmma_kernel<<<g1, 256, 0, stream>>>(bufB, W2t, b2, bufA, N_FACES, 512, 512, 1);
  dim3 g3(256 / 128, N_FACES / 64);
  gemm_bf16_wmma_kernel<<<g3, 256, 0, stream>>>(bufA, W3t, b3, bufB, N_FACES, 256, 512, 1);

  // head 256 -> 12
  head_kernel<<<(N_FACES + 255) / 256, 256, 0, stream>>>(bufB, W4, b4, out12, N_FACES);

  // per-face rotation projection (writes rotations output directly)
  procrustes_kernel<<<(N_FACES + 255) / 256, 256, 0, stream>>>(out12, rot_out, N_FACES);

  // zero accumulators (harness does not re-poison between replays)
  zero_kernel<<<(N_VERTS * 3 + 255) / 256, 256, 0, stream>>>(sums, N_VERTS * 3);
  zero_kernel<<<(N_VERTS + 255) / 256, 256, 0, stream>>>(counts, N_VERTS);

  // transform prisms, write transformed_prism, scatter vertex sums
  transform_scatter_kernel<<<(N_FACES + 255) / 256, 256, 0, stream>>>(
      verts, faces, rot_out, out12, trans_out, sums, counts, N_FACES);

  // vertex mean -> out_features
  finalize_kernel<<<(N_VERTS + 255) / 256, 256, 0, stream>>>(sums, counts, out_features, N_VERTS);
}
